// QuantumHydraSSMCore_83863531421753
// MI455X (gfx1250) — matrix operand order repaired and checked by reading, not compiled, not run on hardware
//
#include <hip/hip_runtime.h>
#include <hip/hip_bf16.h>
#include <math.h>

#define N_QUBITS 6
#define DIM 64
#define BATCH 128
#define SEQ 512
#define QSVT_DEG 3
#define PI_F 3.14159265358979323846f

typedef __attribute__((ext_vector_type(16))) _Float16 v16h;
typedef __attribute__((ext_vector_type(8)))  _Float16 v8h;
typedef __attribute__((ext_vector_type(8)))  float    v8f;
typedef unsigned int u32x4 __attribute__((ext_vector_type(4)));
typedef int          i32x4 __attribute__((ext_vector_type(4)));
typedef int          i32x8 __attribute__((ext_vector_type(8)));

// ---------------------------------------------------------------------------
// ws layout (floats):
//   [0 .. 16384)                 : M matrices. dir d at d*8192: Mr[64*64], Mi[64*64]
//   [16384 .. 16384+2*393216)    : per-direction outputs h[dir][b][t][w]
// ---------------------------------------------------------------------------
#define WS_M_FLOATS   (2 * 2 * DIM * DIM)
#define WS_OUT_BASE   WS_M_FLOATS
#define OUT_ELEMS     (BATCH * SEQ * N_QUBITS)
#define PCSTR         17                 // padded float2 row stride (banks)

struct cpx { float r, i; };
__device__ __forceinline__ cpx cmul(cpx a, cpx b) { return { a.r*b.r - a.i*b.i, a.r*b.i + a.i*b.r }; }
__device__ __forceinline__ cpx cadd(cpx a, cpx b) { return { a.r + b.r, a.i + b.i }; }

// ---------------------------------------------------------------------------
// Setup: build M = V_variational * Q_qsvt (64x64 complex) per direction.
// ---------------------------------------------------------------------------
__global__ void qh_build_M(const float* __restrict__ poly,
                           const float* __restrict__ fwdp,
                           const float* __restrict__ bwdp,
                           float* __restrict__ ws) {
    const int dir = blockIdx.x;
    const float* params = dir ? bwdp : fwdp;
    const int c = threadIdx.x;           // column (input basis state)

    __shared__ float colR[64 * 65];      // +1 pad: avoid bank conflicts
    __shared__ float colI[64 * 65];
    float* er = &colR[c * 65];
    float* ei = &colI[c * 65];

    for (int j = 0; j < DIM; ++j) { er[j] = (j == c) ? 1.f : 0.f; ei[j] = 0.f; }

    auto cnot = [&](int cw, int tw) {
        const int cm = 1 << (5 - cw), tm = 1 << (5 - tw);
        for (int j = 0; j < DIM; ++j) {
            if ((j & cm) && !(j & tm)) {
                int j2 = j | tm;
                float tr = er[j]; er[j] = er[j2]; er[j2] = tr;
                float ti = ei[j]; ei[j] = ei[j2]; ei[j2] = ti;
            }
        }
    };

    for (int d = 0; d <= QSVT_DEG; ++d) {
        const float theta = poly[d] * PI_F;
        for (int j = 0; j < DIM; ++j) {
            float sgn = (float)N_QUBITS - 2.0f * (float)__popc(j);
            float sp, cp; __sincosf(-0.5f * theta * sgn, &sp, &cp);
            cpx v = cmul({ er[j], ei[j] }, { cp, sp });
            er[j] = v.r; ei[j] = v.i;
        }
        for (int i = 0; i < N_QUBITS - 1; ++i) cnot(i, i + 1);
        cnot(N_QUBITS - 1, 0);                      // ring closure
    }

    int idx = 0;
    for (int layer = 0; layer < 2; ++layer) {
        for (int w = 0; w < N_QUBITS; ++w) {
            float tx = params[idx], ty = params[idx + 1], tz = params[idx + 2];
            idx += 3;
            float cx, sx, cy, sy, cz, sz;
            __sincosf(0.5f * tx, &sx, &cx);
            __sincosf(0.5f * ty, &sy, &cy);
            __sincosf(0.5f * tz, &sz, &cz);
            cpx ez  = { cz, -sz }, ezc = { cz, sz };
            cpx Rx00 = { cx, 0 }, Rx01 = { 0, -sx }, Rx10 = { 0, -sx }, Rx11 = { cx, 0 };
            cpx T00 = cadd(cmul({ cy, 0 }, Rx00), cmul({ -sy, 0 }, Rx10));
            cpx T01 = cadd(cmul({ cy, 0 }, Rx01), cmul({ -sy, 0 }, Rx11));
            cpx T10 = cadd(cmul({ sy, 0 }, Rx00), cmul({  cy, 0 }, Rx10));
            cpx T11 = cadd(cmul({ sy, 0 }, Rx01), cmul({  cy, 0 }, Rx11));
            cpx U00 = cmul(ez, T00),  U01 = cmul(ez, T01);
            cpx U10 = cmul(ezc, T10), U11 = cmul(ezc, T11);
            const int m = 1 << (5 - w);
            for (int j0 = 0; j0 < DIM; ++j0) {
                if (j0 & m) continue;
                int j1 = j0 | m;
                cpx a = { er[j0], ei[j0] }, b = { er[j1], ei[j1] };
                cpx na = cadd(cmul(U00, a), cmul(U01, b));
                cpx nb = cadd(cmul(U10, a), cmul(U11, b));
                er[j0] = na.r; ei[j0] = na.i;
                er[j1] = nb.r; ei[j1] = nb.i;
            }
        }
        for (int i = 0; i < N_QUBITS - 1; ++i) cnot(i, i + 1);
    }

    float* Mr = ws + (size_t)dir * 8192;
    float* Mi = Mr + 4096;
    for (int j = 0; j < DIM; ++j) { Mr[j * DIM + c] = er[j]; Mi[j * DIM + c] = ei[j]; }
}

// ---------------------------------------------------------------------------
// compile-time specialized pair reduction: all LDS offsets constant-fold
// ---------------------------------------------------------------------------
template<int BP>
__device__ __forceinline__ void pair_reduce(const float2* __restrict__ pcb,
                                            float& accA, float& accB) {
    constexpr int mk = 1 << BP;
#pragma unroll
    for (int m = 0; m < 32; ++m) {
        const int j0 = ((m >> BP) << (BP + 1)) | (m & (mk - 1));
        const int j1 = j0 | mk;
        float2 v0 = pcb[j0 * PCSTR];
        float2 v1 = pcb[j1 * PCSTR];
        accA += (v0.x * v0.x + v0.y * v0.y) - (v1.x * v1.x + v1.y * v1.y);
        accB += v0.x * v1.x + v0.y * v1.y;
    }
}

// ---------------------------------------------------------------------------
// TDM: DMA one step's 16x6-float angle tile (row stride SEQ*6) into LDS.
// D# built per CDNA5 ISA 8.3/8.4. Issued by one wave; completion via TENSORcnt.
// ---------------------------------------------------------------------------
__device__ __forceinline__ void tdm_load_angles(const float* gsrc, unsigned lds_off) {
    unsigned long long ga = (unsigned long long)(uintptr_t)gsrc;
    u32x4 g0;
    g0[0] = 1u;                                   // count=1 valid descriptor
    g0[1] = lds_off;                              // LDS byte address
    g0[2] = (unsigned)ga;                         // global_addr[31:0]
    g0[3] = (unsigned)(ga >> 32) | (2u << 30);    // global_addr[56:32] | type=2
    i32x8 g1;
    g1[0] = 0x00020000;       // workgroup_mask=0, data_size=2 (4B)
    g1[1] = 6 << 16;          // tensor_dim0 = 6
    g1[2] = 16 << 16;         // tensor_dim1 = 16
    g1[3] = 6 << 16;          // tile_dim0 = 6
    g1[4] = 16;               // tile_dim1 = 16
    g1[5] = SEQ * N_QUBITS;   // tensor_dim0_stride = 3072 elements
    g1[6] = 0;
    g1[7] = 0;
    i32x4 z4 = { 0, 0, 0, 0 };
#if __clang_major__ >= 23
    i32x8 z8 = { 0, 0, 0, 0, 0, 0, 0, 0 };
    __builtin_amdgcn_tensor_load_to_lds(g0, g1, z4, z4, z8, 0);
#else
    __builtin_amdgcn_tensor_load_to_lds(g0, g1, z4, z4, 0);
#endif
}

// ---------------------------------------------------------------------------
// Main scan kernel. Grid = 16 blocks (dir*8 + batch-tile), 128 threads (4 waves).
// ---------------------------------------------------------------------------
struct SharedBlk {
    alignas(32) _Float16 psi0T[16 * DIM];   // [b][k], contiguous K for B frags
    alignas(16) float2   pc[DIM * PCSTR];   // psi1 (re,im) [row][b], padded rows
    float hB[16 * N_QUBITS];                // hidden angles
    alignas(16) float xbuf[2][16 * N_QUBITS]; // TDM double buffer of x angles
};

__global__ void __launch_bounds__(128)
qh_scan(const float* __restrict__ angles, const float* __restrict__ ws_m,
        float* __restrict__ ws_out) {
    __shared__ SharedBlk sh;

    const int tid   = threadIdx.x;
    const int dir   = blockIdx.x >> 3;
    const int bBase = (blockIdx.x & 7) * 16;
    const int lane  = tid & 31;
    const int wave  = tid >> 5;
    const int g     = lane >> 4;     // lane half group
    const int lm    = lane & 15;
    const int R     = wave * 16;     // row base of this wave's D tile

    // psi0-build ownership: thread -> (batch be, k-octet m8)
    const int be = tid >> 3;
    const int m8 = tid & 7;

    // reduction ownership: thread -> (batch rb, wire rw) for tid < 96
    const int rb = tid & 15;
    const int rw = tid >> 4;

    const float* angBase = angles + (size_t)bBase * SEQ * N_QUBITS;

    // ---- hoist fixed A fragments (rows R..R+15 of Mr / Mi) into VGPRs ----
    const float* Mr = ws_m + (size_t)dir * 8192;
    const float* Mi = Mr + 4096;
    v16h Ar[2], Ai[2];
    {
        alignas(32) _Float16 tr[16], ti[16];
        const int row = R + lm;
        for (int q = 0; q < 2; ++q) {
            for (int h = 0; h < 16; ++h) {
                // 16-bit A-matrix 16x32: contiguous K per group (matches B)
                int k = 32 * q + 16 * g + h;
                tr[h] = (_Float16)Mr[row * DIM + k];
                ti[h] = (_Float16)Mi[row * DIM + k];
            }
            Ar[q] = *(const v16h*)tr;
            Ai[q] = *(const v16h*)ti;
        }
    }

    // ---- h0 = 0; prologue TDM for step 0's angles ----
    if (tid < 96) sh.hB[tid] = 0.f;
    if (wave == 3) {
        int tp0 = dir ? (SEQ - 1) : 0;
        tdm_load_angles(angBase + (size_t)tp0 * N_QUBITS,
                        (unsigned)(uintptr_t)&sh.xbuf[0][0]);
        __builtin_amdgcn_s_wait_tensorcnt(0);
    }
    __syncthreads();

    for (int t = 0; t < SEQ; ++t) {
        const int tpos = dir ? (SEQ - 1 - t) : t;   // bwd pass walks reversed

        // wave 3: kick off DMA of next step's angle tile (overlaps whole step)
        if (wave == 3 && (t + 1) < SEQ) {
            int tpn = dir ? (SEQ - 2 - t) : (t + 1);
            tdm_load_angles(angBase + (size_t)tpn * N_QUBITS,
                            (unsigned)(uintptr_t)&sh.xbuf[(t + 1) & 1][0]);
        }

        // ---- psi0[b][k] = prod_w (bit_{5-w}(k) ? sin : cos)(h_bw/2) (real) ----
        {
            float c0, s0, c1, s1, c2, s2, c3, s3, c4, s4, c5, s5;
            const float* hb = &sh.hB[be * N_QUBITS];
            __sincosf(0.5f * hb[0], &s0, &c0);
            __sincosf(0.5f * hb[1], &s1, &c1);
            __sincosf(0.5f * hb[2], &s2, &c2);
            __sincosf(0.5f * hb[3], &s3, &c3);
            __sincosf(0.5f * hb[4], &s4, &c4);
            __sincosf(0.5f * hb[5], &s5, &c5);
            float pre = ((m8 & 4) ? s0 : c0) * ((m8 & 2) ? s1 : c1)
                      * ((m8 & 1) ? s2 : c2);
            float a0 = pre * c3, a1 = pre * s3;
            float b00 = a0 * c4, b01 = a0 * s4, b10 = a1 * c4, b11 = a1 * s4;
            v8h oct;
            oct[0] = (_Float16)(b00 * c5); oct[1] = (_Float16)(b00 * s5);
            oct[2] = (_Float16)(b01 * c5); oct[3] = (_Float16)(b01 * s5);
            oct[4] = (_Float16)(b10 * c5); oct[5] = (_Float16)(b10 * s5);
            oct[6] = (_Float16)(b11 * c5); oct[7] = (_Float16)(b11 * s5);
            *(v8h*)(&sh.psi0T[be * DIM + m8 * 8]) = oct;
        }
        __syncthreads();

        // ---- psi1 = M * psi0 : two real GEMMs on the matrix core ----
        v8f dr = {}, di = {};
#pragma unroll
        for (int q = 0; q < 2; ++q) {
            // 16-bit B-matrix 32x16 layout: lane=col, contiguous K per group
            v16h B = *(const v16h*)(&sh.psi0T[lm * DIM + 32 * q + 16 * g]);
            dr = __builtin_amdgcn_wmma_f32_16x16x32_f16(false, Ar[q], false, B,
                                                        (short)0, dr, false, false);
            di = __builtin_amdgcn_wmma_f32_16x16x32_f16(false, Ai[q], false, B,
                                                        (short)0, di, false, false);
        }
#pragma unroll
        for (int r = 0; r < 8; ++r) {          // C/D layout: VGPR r -> row r+8g
            int row = R + r + 8 * g;
            sh.pc[row * PCSTR + lm] = make_float2(dr[r], di[r]);
        }
        __syncthreads();

        // ---- measurement with RY(x) folded in:
        //  <Z_w> = cos(x)*sum_j sign_w(j)|psi_j|^2 - sin(x)*2*sum_pairs Re(p0*conj p1)
        if (tid < 96) {
            float accA = 0.f, accB = 0.f;
            const float2* pcb = &sh.pc[rb];
            switch (rw) {                       // compile-time masks: all LDS
                case 0: pair_reduce<5>(pcb, accA, accB); break;  // offsets fold
                case 1: pair_reduce<4>(pcb, accA, accB); break;
                case 2: pair_reduce<3>(pcb, accA, accB); break;
                case 3: pair_reduce<2>(pcb, accA, accB); break;
                case 4: pair_reduce<1>(pcb, accA, accB); break;
                default: pair_reduce<0>(pcb, accA, accB); break;
            }
            float xv = sh.xbuf[t & 1][rb * N_QUBITS + rw];
            float sx_, cx_; __sincosf(xv, &sx_, &cx_);
            float hn = cx_ * accA - sx_ * (2.f * accB);
            sh.hB[rb * N_QUBITS + rw] = hn;
            ws_out[(size_t)dir * OUT_ELEMS +
                   ((size_t)(bBase + rb) * SEQ + tpos) * N_QUBITS + rw] = hn;
        }
        if (wave == 3) __builtin_amdgcn_s_wait_tensorcnt(0);  // next tile landed
        __syncthreads();
    }
}

// ---------------------------------------------------------------------------
// Combine: out = sigmoid(fc)*h_fwd + sigmoid(bc)*h_bwd
// ---------------------------------------------------------------------------
__global__ void qh_combine(const float* __restrict__ ws_out,
                           const float* __restrict__ fc,
                           const float* __restrict__ bc,
                           float* __restrict__ out) {
    int i = blockIdx.x * blockDim.x + threadIdx.x;
    if (i >= OUT_ELEMS) return;
    float sf = 1.f / (1.f + __expf(-fc[0]));
    float sb = 1.f / (1.f + __expf(-bc[0]));
    out[i] = sf * ws_out[i] + sb * ws_out[OUT_ELEMS + i];
}

extern "C" void kernel_launch(void* const* d_in, const int* in_sizes, int n_in,
                              void* d_out, int out_size, void* d_ws, size_t ws_size,
                              hipStream_t stream) {
    (void)in_sizes; (void)n_in; (void)out_size; (void)ws_size;
    const float* angles = (const float*)d_in[0];
    const float* poly   = (const float*)d_in[1];
    const float* fwdp   = (const float*)d_in[2];
    const float* bwdp   = (const float*)d_in[3];
    const float* fcoef  = (const float*)d_in[4];
    const float* bcoef  = (const float*)d_in[5];
    float* ws = (float*)d_ws;
    float* out = (float*)d_out;

    qh_build_M<<<2, 64, 0, stream>>>(poly, fwdp, bwdp, ws);
    qh_scan<<<16, 128, 0, stream>>>(angles, ws, ws + WS_OUT_BASE);
    qh_combine<<<(OUT_ELEMS + 255) / 256, 256, 0, stream>>>(ws + WS_OUT_BASE,
                                                            fcoef, bcoef, out);
}
